// MultiHeadAttention_37752762532293
// MI455X (gfx1250) — matrix operand order repaired and checked by reading, not compiled, not run on hardware
//
#include <hip/hip_runtime.h>
#include <stdint.h>

typedef _Float16 half_t;
typedef __attribute__((ext_vector_type(8)))  _Float16 v8h;
typedef __attribute__((ext_vector_type(16))) _Float16 v16h;
typedef __attribute__((ext_vector_type(8)))  float    v8f;
typedef __attribute__((ext_vector_type(4)))  unsigned int u32x4;
typedef __attribute__((ext_vector_type(8)))  int      i32x8;
typedef __attribute__((ext_vector_type(4)))  int      i32x4;

#define DIMN 1024
#define SEQ  2048
#define NBAT 4
#define NH   16
#define HDIM 64
#define NM   (NBAT*SEQ)   // 8192 rows

// ---------------------------------------------------------------------------
// WMMA helper: D = A(16x32 f16) * B(32x16 f16) + C(16x16 f32)
// ---------------------------------------------------------------------------
static __device__ __forceinline__ v8f wmma_f16(v16h a, v16h b, v8f c) {
  return __builtin_amdgcn_wmma_f32_16x16x32_f16(false, a, false, b,
                                                (short)0, c, false, false);
}

// A-operand slice, contiguous K order (matches B fills):
//  lanes 0-15 (row m): K = k+0..15 ; lanes 16-31: k+16..31
static __device__ __forceinline__ v16h load_a16(const half_t* row, int k, int hi) {
  const v8h lo = *(const v8h*)(row + k + hi*16);
  const v8h hh = *(const v8h*)(row + k + hi*16 + 8);
  return __builtin_shufflevector(lo, hh, 0,1,2,3,4,5,6,7,8,9,10,11,12,13,14,15);
}

// ---------------------------------------------------------------------------
// Tensor Data Mover: DMA a 2-D tile (data_size = 8B units) into LDS.
// D# bitfields packed per CDNA5 ISA ch.8 (group0: count/lds/global/type,
// group1: data_size, tensor dims, tile dims, dim0 stride). 1-D copies use
// tile_d1 = 0 (unused). 6-arg builtin (clang-23): groups 2/3 and the extra
// group are zero (tensors <= 2-D), cpol = 0.
// ---------------------------------------------------------------------------
static __device__ __forceinline__ void tdm_load_2d(
    uint32_t lds_byte_off, const void* gsrc,
    uint32_t tensor_d0, uint32_t tensor_d1,
    uint32_t tile_d0, uint32_t tile_d1, uint64_t stride0_units) {
  const uint64_t ga = (uint64_t)(uintptr_t)gsrc;
  u32x4 g0;
  g0[0] = 1u;                                                // count = 1
  g0[1] = lds_byte_off;                                      // lds_addr
  g0[2] = (uint32_t)ga;                                      // global_addr[31:0]
  g0[3] = (uint32_t)((ga >> 32) & 0x01ffffffu) | (2u << 30); // addr[56:32]|type=2
  i32x8 g1;
  g1[0] = (int)(3u << 16);                                   // data_size = 8B
  g1[1] = (int)((tensor_d0 & 0xffffu) << 16);                // tensor_dim0[15:0]
  g1[2] = (int)((tensor_d0 >> 16) | ((tensor_d1 & 0xffffu) << 16));
  g1[3] = (int)((tensor_d1 >> 16) | (tile_d0 << 16));        // tile_dim0
  g1[4] = (int)(tile_d1 & 0xffffu);                          // tile_dim1 (0=1D)
  g1[5] = (int)(uint32_t)stride0_units;                      // stride0[31:0]
  g1[6] = (int)(uint32_t)((stride0_units >> 32) & 0xffffu);  // stride0[47:32]
  g1[7] = 0;
  const i32x4 z4 = {0, 0, 0, 0};
  const i32x8 z8 = {0, 0, 0, 0, 0, 0, 0, 0};
  __builtin_amdgcn_tensor_load_to_lds(g0, g1, z4, z4, z8, 0);
}

// ---------------------------------------------------------------------------
// fp32 -> f16 conversion (8 elems/thread, b128 loads)
// ---------------------------------------------------------------------------
__global__ void cvt_f32_to_f16(const float* __restrict__ in,
                               half_t* __restrict__ out, int n) {
  int i = (blockIdx.x*blockDim.x + threadIdx.x)*8;
  if (i >= n) return;
  v8f x = *(const v8f*)(in + i);
  v8h y;
  #pragma unroll
  for (int j=0;j<8;++j) y[j] = (half_t)x[j];
  *(v8h*)(out + i) = y;
}

// ---------------------------------------------------------------------------
// Projection GEMM: out[m,n] = sum_k A[m,k]*W[n,k] + bias[n].
// Block = 8 waves, 128x128 tile; wave = 32x64. Operands direct from L2
// (entire W is L2-resident on a 192MB L2).
// vmode 0: store [b,h,s,hd] (Q,K);  vmode 1: store [b,h,hd,s] (V^T).
// ---------------------------------------------------------------------------
__global__ __launch_bounds__(256) void proj_gemm(
    const half_t* __restrict__ A, const half_t* __restrict__ W,
    const float* __restrict__ bias, half_t* __restrict__ out, int vmode) {
  const int tid  = threadIdx.x;
  const int wave = tid >> 5;
  const int lane = tid & 31;
  const int ln   = lane & 15;
  const int hi   = lane >> 4;
  const int wm   = wave & 3;
  const int wn   = wave >> 2;
  const int m_base = blockIdx.x*128 + wm*32;
  const int n_base = blockIdx.y*128 + wn*64;

  const half_t* ar0 = A + (size_t)(m_base + ln)*DIMN;
  const half_t* ar1 = A + (size_t)(m_base + 16 + ln)*DIMN;

  v8f acc[2][4] = {};
  for (int k = 0; k < DIMN; k += 32) {
    __builtin_prefetch(ar0 + k + 128, 0, 1);   // global_prefetch_b8
    v16h a0 = load_a16(ar0, k, hi);
    v16h a1 = load_a16(ar1, k, hi);
    #pragma unroll
    for (int nt = 0; nt < 4; ++nt) {
      const int col = n_base + nt*16 + ln;
      v16h b = *(const v16h*)(W + (size_t)col*DIMN + k + hi*16);
      acc[0][nt] = wmma_f16(a0, b, acc[0][nt]);
      acc[1][nt] = wmma_f16(a1, b, acc[1][nt]);
    }
  }
  #pragma unroll
  for (int mt = 0; mt < 2; ++mt)
  #pragma unroll
  for (int nt = 0; nt < 4; ++nt) {
    const int col = n_base + nt*16 + ln;
    const float bvv = bias[col];
    const int h = col >> 6, hd = col & 63;
    #pragma unroll
    for (int r = 0; r < 8; ++r) {
      const int row = m_base + mt*16 + r + 8*hi;   // C layout: M=r / r+8
      const int bb = row >> 11, s = row & (SEQ-1);
      const float v = acc[mt][nt][r] + bvv;
      size_t idx;
      if (vmode) idx = ((size_t)(bb*NH + h)*HDIM + hd)*SEQ + s;
      else       idx = ((size_t)(bb*NH + h)*SEQ  + s)*HDIM + hd;
      out[idx] = (half_t)v;
    }
  }
}

// ---------------------------------------------------------------------------
// Flash attention. grid = (B*H, S/128), block = 8 waves (16 queries each).
// All waves share (b,h) => K/V tiles are staged ONCE per block into LDS by
// the Tensor Data Mover (double-buffered, overlapped with compute), instead
// of 8x redundant L2 reads. Softmax runs in the exp2 domain.
// ---------------------------------------------------------------------------
__global__ __launch_bounds__(256) void flash_attn(
    const half_t* __restrict__ Qh, const half_t* __restrict__ Kh,
    const half_t* __restrict__ Vt, half_t* __restrict__ O) {
  __shared__ half_t kst[2][32][HDIM];   // key tiles   (2 x 4KB)
  __shared__ half_t vst[2][HDIM][32];   // V^T tiles   (2 x 4KB)
  __shared__ half_t plds[8][16][32];    // per-wave P re-layout (8KB)

  const int tid = threadIdx.x, wave = tid>>5, lane = tid&31;
  const int ln = lane & 15, hi = lane >> 4;
  const int bh = blockIdx.x;
  const int q0 = blockIdx.y*128 + wave*16;
  const half_t* Qb = Qh + (size_t)bh*SEQ*HDIM;
  const half_t* Kb = Kh + (size_t)bh*SEQ*HDIM;
  const half_t* Vb = Vt + (size_t)bh*HDIM*SEQ;

  const half_t* qrow = Qb + (size_t)(q0 + ln)*HDIM;
  const v16h aq0 = load_a16(qrow, 0,  hi);
  const v16h aq1 = load_a16(qrow, 32, hi);

  v8f o0 = {}, o1 = {}, o2 = {}, o3 = {};
  float mrun[8], lrun[8];
  #pragma unroll
  for (int r = 0; r < 8; ++r) { mrun[r] = -1e30f; lrun[r] = 0.f; }

  // Prologue: TDM-stage tile 0 (K: 4KB contiguous; V^T: 64 rows x 64B,
  // row stride 4KB -> 512 x 8B units).
  if (wave == 0) {
    tdm_load_2d((uint32_t)(uintptr_t)&kst[0][0][0], Kb, 512, 0, 512, 0, 0);
    tdm_load_2d((uint32_t)(uintptr_t)&vst[0][0][0], Vb, 512, HDIM, 8, HDIM, 512);
  }

  for (int kk = 0; kk < SEQ; kk += 32) {
    const int buf = (kk >> 5) & 1;
    if (wave == 0) __builtin_amdgcn_s_wait_tensorcnt(0);
    __syncthreads();                       // tiles[buf] visible; tiles[buf^1] free
    if (wave == 0 && kk + 32 < SEQ) {      // prefetch next tile via TDM
      const int nb = buf ^ 1;
      tdm_load_2d((uint32_t)(uintptr_t)&kst[nb][0][0],
                  Kb + (size_t)(kk + 32)*HDIM, 512, 0, 512, 0, 0);
      tdm_load_2d((uint32_t)(uintptr_t)&vst[nb][0][0],
                  Vb + (kk + 32), 512, HDIM, 8, HDIM, 512);
    }

    v8f c0 = {}, c1 = {};
    {
      const half_t* k0 = &kst[buf][ln][hi*16];
      const half_t* k1 = &kst[buf][16 + ln][hi*16];
      c0 = wmma_f16(aq0, *(const v16h*)(k0),      c0);
      c0 = wmma_f16(aq1, *(const v16h*)(k0 + 32), c0);
      c1 = wmma_f16(aq0, *(const v16h*)(k1),      c1);
      c1 = wmma_f16(aq1, *(const v16h*)(k1 + 32), c1);
    }
    const float SCL = 0.125f * 1.44269504088896f;  // HD^-0.5 * log2(e)
    #pragma unroll
    for (int r = 0; r < 8; ++r) {
      float s0 = c0[r]*SCL, s1 = c1[r]*SCL;        // exp2 domain
      float mx = fmaxf(s0, s1);
      #pragma unroll
      for (int off = 1; off < 16; off <<= 1)
        mx = fmaxf(mx, __shfl_xor(mx, off, 32));   // half-wave row reduce
      const float mnew  = fmaxf(mrun[r], mx);
      const float alpha = exp2f(mrun[r] - mnew);
      const float p0 = exp2f(s0 - mnew), p1 = exp2f(s1 - mnew);
      float rs = p0 + p1;
      #pragma unroll
      for (int off = 1; off < 16; off <<= 1) rs += __shfl_xor(rs, off, 32);
      lrun[r] = lrun[r]*alpha + rs;
      mrun[r] = mnew;
      o0[r] *= alpha; o1[r] *= alpha; o2[r] *= alpha; o3[r] *= alpha;
      plds[wave][r + 8*hi][ln]      = (half_t)p0;  // C->A transpose via LDS
      plds[wave][r + 8*hi][16 + ln] = (half_t)p1;
    }
    __syncthreads();                     // plds write -> read (also paces waves)
    const v16h ap = load_a16(&plds[wave][ln][0], 0, hi);
    {
      const half_t* v0 = &vst[buf][ln][hi*16];
      o0 = wmma_f16(ap, *(const v16h*)(v0),           o0);
      o1 = wmma_f16(ap, *(const v16h*)(v0 + 16*32),   o1);
      o2 = wmma_f16(ap, *(const v16h*)(v0 + 32*32),   o2);
      o3 = wmma_f16(ap, *(const v16h*)(v0 + 48*32),   o3);
    }
    // no trailing barrier: next iteration's top barrier sequences reuse
  }

  const int b = bh >> 4, h = bh & 15;
  #pragma unroll
  for (int r = 0; r < 8; ++r) {
    const float inv = 1.0f / lrun[r];
    const int row = q0 + r + 8*hi;
    half_t* orow = O + (size_t)(b*SEQ + row)*DIMN + h*HDIM;
    orow[ 0 + ln] = (half_t)(o0[r]*inv);
    orow[16 + ln] = (half_t)(o1[r]*inv);
    orow[32 + ln] = (half_t)(o2[r]*inv);
    orow[48 + ln] = (half_t)(o3[r]*inv);
  }
}

// ---------------------------------------------------------------------------
// Output projection + fused LayerNorm. One block = 16 full rows (8 waves x
// 128 cols each). Cross-wave mean/var reduction via LDS, fp32 output.
// ---------------------------------------------------------------------------
__global__ __launch_bounds__(256) void oproj_ln(
    const half_t* __restrict__ X, const half_t* __restrict__ Wo,
    const float* __restrict__ bo, const float* __restrict__ gamma,
    const float* __restrict__ beta, float* __restrict__ out) {
  __shared__ float red1[8][2][8];
  __shared__ float red2[8][2][8];
  __shared__ float mu_s[16], rs_s[16];
  const int tid = threadIdx.x, wave = tid>>5, lane = tid&31;
  const int ln = lane & 15, hi = lane >> 4;
  const int m0 = blockIdx.x*16;
  const half_t* xrow = X + (size_t)(m0 + ln)*DIMN;

  v8f acc[8] = {};
  for (int k = 0; k < DIMN; k += 32) {
    v16h a = load_a16(xrow, k, hi);
    #pragma unroll
    for (int nt = 0; nt < 8; ++nt) {
      const int col = wave*128 + nt*16 + ln;
      v16h b = *(const v16h*)(Wo + (size_t)col*DIMN + k + hi*16);
      acc[nt] = wmma_f16(a, b, acc[nt]);
    }
  }

  float s1[8] = {}, s2[8] = {};
  #pragma unroll
  for (int nt = 0; nt < 8; ++nt) {
    const float bvv = bo[wave*128 + nt*16 + ln];
    #pragma unroll
    for (int r = 0; r < 8; ++r) {
      const float v = acc[nt][r] + bvv;
      acc[nt][r] = v;
      s1[r] += v; s2[r] += v*v;
    }
  }
  #pragma unroll
  for (int r = 0; r < 8; ++r) {
    #pragma unroll
    for (int off = 1; off < 16; off <<= 1) {
      s1[r] += __shfl_xor(s1[r], off, 32);
      s2[r] += __shfl_xor(s2[r], off, 32);
    }
    if (ln == 0) { red1[wave][hi][r] = s1[r]; red2[wave][hi][r] = s2[r]; }
  }
  __syncthreads();
  if (tid < 16) {
    const int r = tid & 7, hh = tid >> 3;     // row = r + 8*hh == tid
    float S1 = 0.f, S2 = 0.f;
    #pragma unroll
    for (int w = 0; w < 8; ++w) { S1 += red1[w][hh][r]; S2 += red2[w][hh][r]; }
    const float mean = S1 * (1.0f/DIMN);
    const float var  = S2 * (1.0f/DIMN) - mean*mean;
    mu_s[tid] = mean;
    rs_s[tid] = rsqrtf(var + 1e-5f);
  }
  __syncthreads();
  #pragma unroll
  for (int nt = 0; nt < 8; ++nt) {
    const int col = wave*128 + nt*16 + ln;
    const float g = gamma[col], bt = beta[col];
    #pragma unroll
    for (int r = 0; r < 8; ++r) {
      const int row = r + 8*hi;
      out[(size_t)(m0 + row)*DIMN + col] =
          (acc[nt][r] - mu_s[row])*rs_s[row]*g + bt;
    }
  }
}

// ---------------------------------------------------------------------------
extern "C" void kernel_launch(void* const* d_in, const int* in_sizes, int n_in,
                              void* d_out, int out_size, void* d_ws, size_t ws_size,
                              hipStream_t stream) {
  const float* q   = (const float*)d_in[0];
  const float* k   = (const float*)d_in[1];
  const float* v   = (const float*)d_in[2];
  const float* Wq  = (const float*)d_in[3];
  const float* bq  = (const float*)d_in[4];
  const float* Wk  = (const float*)d_in[5];
  const float* bk  = (const float*)d_in[6];
  const float* Wv  = (const float*)d_in[7];
  const float* bv  = (const float*)d_in[8];
  const float* Wo  = (const float*)d_in[9];
  const float* bo  = (const float*)d_in[10];
  const float* gamma = (const float*)d_in[11];
  const float* beta  = (const float*)d_in[12];
  float* out = (float*)d_out;

  // Workspace (f16 halves): 3*8Mi + 4*1Mi + 3*8Mi = 52Mi halves = 104 MB.
  // Attention output reuses the (dead) q-f16 buffer.
  half_t* ws = (half_t*)d_ws;
  const size_t SX = (size_t)NM*DIMN;      // 8 Mi halves per activation tensor
  const size_t SW = (size_t)DIMN*DIMN;    // 1 Mi halves per weight
  half_t* qf   = ws;
  half_t* kf   = qf + SX;
  half_t* vf   = kf + SX;
  half_t* wq16 = vf + SX;
  half_t* wk16 = wq16 + SW;
  half_t* wv16 = wk16 + SW;
  half_t* wo16 = wv16 + SW;
  half_t* Qh   = wo16 + SW;
  half_t* Kh   = Qh + SX;
  half_t* Vt   = Kh + SX;
  half_t* attn = qf;                      // reuse: qf dead after Q projection

  const int CT = 256;
  cvt_f32_to_f16<<<(int)(SX/(CT*8)), CT, 0, stream>>>(q,  qf,   (int)SX);
  cvt_f32_to_f16<<<(int)(SX/(CT*8)), CT, 0, stream>>>(k,  kf,   (int)SX);
  cvt_f32_to_f16<<<(int)(SX/(CT*8)), CT, 0, stream>>>(v,  vf,   (int)SX);
  cvt_f32_to_f16<<<(int)(SW/(CT*8)), CT, 0, stream>>>(Wq, wq16, (int)SW);
  cvt_f32_to_f16<<<(int)(SW/(CT*8)), CT, 0, stream>>>(Wk, wk16, (int)SW);
  cvt_f32_to_f16<<<(int)(SW/(CT*8)), CT, 0, stream>>>(Wv, wv16, (int)SW);
  cvt_f32_to_f16<<<(int)(SW/(CT*8)), CT, 0, stream>>>(Wo, wo16, (int)SW);

  dim3 pg(NM/128, DIMN/128);              // 64 x 8
  proj_gemm<<<pg, 256, 0, stream>>>(qf, wq16, bq, Qh, 0);
  proj_gemm<<<pg, 256, 0, stream>>>(kf, wk16, bk, Kh, 0);
  proj_gemm<<<pg, 256, 0, stream>>>(vf, wv16, bv, Vt, 1);   // V transposed

  flash_attn<<<dim3(NBAT*NH, SEQ/128), 256, 0, stream>>>(Qh, Kh, Vt, attn);

  oproj_ln<<<NM/16, 256, 0, stream>>>(attn, wo16, bo, gamma, beta, out);
}
